// VimBlock_87213605913198
// MI455X (gfx1250) — matrix operand order repaired and checked by reading, not compiled, not run on hardware
//
#include <hip/hip_runtime.h>
#include <hip/hip_bf16.h>

#define BATCH   4
#define SEQ     1024
#define DM      384          // D_MODEL
#define DI      768          // D_INNER
#define DSTATE  16
#define DTRANK  24
#define TOK     (BATCH*SEQ)  // 4096 tokens

typedef __bf16 bf16x16 __attribute__((ext_vector_type(16)));
typedef __bf16 bf16x8  __attribute__((ext_vector_type(8)));
typedef float  f32x8   __attribute__((ext_vector_type(8)));

__device__ __forceinline__ unsigned short f2bf(float f) {
    unsigned int u = __float_as_uint(f);
    unsigned int r = (u + 0x7FFFu + ((u >> 16) & 1u)) >> 16;
    return (unsigned short)r;
}

__device__ __forceinline__ float silu_f(float v) {
    return v / (1.f + __expf(-v));
}

// ---------------------------------------------------------------------------
// Weight convert + transpose: W[K][N] f32  ->  WT[Npad][Kpad] bf16 (zero pad)
// ---------------------------------------------------------------------------
__global__ void convtrans_kernel(const float* __restrict__ W,
                                 unsigned short* __restrict__ WT,
                                 int K, int N, int Kpad, int Npad) {
    size_t idx = (size_t)blockIdx.x * blockDim.x + threadIdx.x;
    size_t total = (size_t)Kpad * Npad;
    if (idx >= total) return;
    int n = (int)(idx / Kpad);
    int k = (int)(idx % Kpad);
    float v = (k < K && n < N) ? W[(size_t)k * N + n] : 0.f;
    WT[idx] = f2bf(v);
}

// ---------------------------------------------------------------------------
// LayerNorm over DM per token -> bf16 activations
// ---------------------------------------------------------------------------
__global__ void ln_kernel(const float* __restrict__ x,
                          const float* __restrict__ g,
                          const float* __restrict__ b,
                          unsigned short* __restrict__ xn_bf) {
    __shared__ float red0[128];
    __shared__ float red1[128];
    int t = blockIdx.x;           // token
    int tid = threadIdx.x;
    const float* xr = x + (size_t)t * DM;
    float s = 0.f, s2 = 0.f;
    for (int i = tid; i < DM; i += 128) { float v = xr[i]; s += v; s2 += v * v; }
    red0[tid] = s; red1[tid] = s2;
    __syncthreads();
    for (int off = 64; off > 0; off >>= 1) {
        if (tid < off) { red0[tid] += red0[tid + off]; red1[tid] += red1[tid + off]; }
        __syncthreads();
    }
    float mu  = red0[0] * (1.f / DM);
    float var = red1[0] * (1.f / DM) - mu * mu;
    float rs  = rsqrtf(var + 1e-5f);
    for (int i = tid; i < DM; i += 128) {
        float v = (xr[i] - mu) * rs * g[i] + b[i];
        xn_bf[(size_t)t * DM + i] = f2bf(v);
    }
}

// ---------------------------------------------------------------------------
// WMMA GEMM:  C[M][N] f32 = A[M][K]bf16 * BT[N][K]bf16 (+bias)(+residual)
//
// Block = 128 threads = 4 waves, macro-tile 64(M) x 64(N).
// Each wave computes a 16x64 strip: one A fragment is reused across four
// back-to-back v_wmma_f32_16x16x32_bf16 against four B fragments, so each
// K-step issues 10 b128 loads for 4 WMMAs (4x the compute density of a
// single-tile wave). Operands are L2-resident (192MB L2), so direct global
// b128 loads + prefetch are the right data path; no LDS staging needed.
// ---------------------------------------------------------------------------
__global__ __launch_bounds__(128)
void gemm_wmma_bf16(const unsigned short* __restrict__ A,
                    const unsigned short* __restrict__ BT,
                    const float* __restrict__ bias,
                    const float* __restrict__ residual,
                    float* __restrict__ C,
                    int M, int N, int K) {
    const int wave   = threadIdx.x >> 5;
    const int lane   = threadIdx.x & 31;          // wave32
    const int tile_m = (blockIdx.y * 4 + wave) * 16;
    const int tile_n = blockIdx.x * 64;

    const unsigned short* Arow = A  + (size_t)(tile_m + (lane & 15)) * K;
    const unsigned short* Brow = BT + (size_t)(tile_n + (lane & 15)) * K;
    const int aoff = (lane < 16) ? 0 : 16;        // A: lanes16-31 hold K half +16
    const int boff = (lane < 16) ? 0 : 16;        // B: lanes16-31 hold K half +16
    const size_t bstride = (size_t)16 * K;        // next 16-column B sub-tile

    f32x8 acc0 = {}, acc1 = {}, acc2 = {}, acc3 = {};

    for (int k0 = 0; k0 < K; k0 += 32) {
        __builtin_prefetch((const void*)(Arow + k0 + 64), 0, 1);

        // A fragment: 16x32 bf16, elems 0..7 = K(aoff..+7), elems 8..15 = +8
        bf16x8 alo = *reinterpret_cast<const bf16x8*>(Arow + k0 + aoff);
        bf16x8 ahi = *reinterpret_cast<const bf16x8*>(Arow + k0 + aoff + 8);
        bf16x16 a;
        #pragma unroll
        for (int i = 0; i < 8; ++i) { a[i] = alo[i]; a[i + 8] = ahi[i]; }

        f32x8* accs[4] = { &acc0, &acc1, &acc2, &acc3 };
        #pragma unroll
        for (int j = 0; j < 4; ++j) {
            const unsigned short* Bp = Brow + (size_t)j * bstride + k0 + boff;
            bf16x8 blo = *reinterpret_cast<const bf16x8*>(Bp);
            bf16x8 bhi = *reinterpret_cast<const bf16x8*>(Bp + 8);
            bf16x16 b;
            #pragma unroll
            for (int i = 0; i < 8; ++i) { b[i] = blo[i]; b[i + 8] = bhi[i]; }
            *accs[j] = __builtin_amdgcn_wmma_f32_16x16x32_bf16(
                false, a, false, b, (short)0, *accs[j], false, false);
        }
    }

    // C/D layout: VGPR r -> M=r (lanes 0-15) / M=r+8 (lanes 16-31), N=lane&15
    const int rbase = tile_m + ((lane < 16) ? 0 : 8);
    f32x8* accs[4] = { &acc0, &acc1, &acc2, &acc3 };
    #pragma unroll
    for (int j = 0; j < 4; ++j) {
        const int col = tile_n + j * 16 + (lane & 15);
        const float bv = bias ? bias[col] : 0.f;
        #pragma unroll
        for (int r = 0; r < 8; ++r) {
            float v = (*accs[j])[r] + bv;
            size_t o = (size_t)(rbase + r) * N + col;
            if (residual) v += residual[o];
            C[o] = v;
        }
    }
}

// ---------------------------------------------------------------------------
// Split xz -> xb bf16 (first half); zb stays in xz for the gate
// ---------------------------------------------------------------------------
__global__ void split_kernel(const float* __restrict__ xz,
                             unsigned short* __restrict__ xb_bf) {
    size_t idx = (size_t)blockIdx.x * blockDim.x + threadIdx.x;
    if (idx >= (size_t)TOK * DM) return;
    size_t t = idx / DM;
    int c = (int)(idx % DM);
    xb_bf[idx] = f2bf(xz[t * (2 * DM) + c]);
}

// ---------------------------------------------------------------------------
// Depthwise conv(4) + SiLU. dir=0: causal (left pad); dir=1: anti-causal,
// taps reversed (== flip-conv-flip of the reference backward branch).
// ---------------------------------------------------------------------------
__global__ void conv_silu_kernel(const float* __restrict__ upre,
                                 const float* __restrict__ conv_w,
                                 const float* __restrict__ conv_b,
                                 float* __restrict__ u,
                                 unsigned short* __restrict__ u_bf,
                                 int dir) {
    size_t idx = (size_t)blockIdx.x * blockDim.x + threadIdx.x;
    if (idx >= (size_t)TOK * DI) return;
    int c = (int)(idx % DI);
    size_t tok = idx / DI;
    int t = (int)(tok % SEQ);
    int b = (int)(tok / SEQ);
    float acc = conv_b[c];
    #pragma unroll
    for (int k = 0; k < 4; ++k) {
        int tp = (dir == 0) ? (t - 3 + k) : (t + 3 - k);
        if (tp >= 0 && tp < SEQ)
            acc += conv_w[c * 4 + k] * upre[((size_t)b * SEQ + tp) * DI + c];
    }
    float s = silu_f(acc);
    u[idx]   = s;
    u_bf[idx] = f2bf(s);
}

// ---------------------------------------------------------------------------
// dtr (first 24 of x_dbl) -> bf16 padded to K=32
// ---------------------------------------------------------------------------
__global__ void dtr_kernel(const float* __restrict__ xdbl,
                           unsigned short* __restrict__ dtr_bf) {
    size_t idx = (size_t)blockIdx.x * blockDim.x + threadIdx.x;
    if (idx >= (size_t)TOK * 32) return;
    size_t t = idx / 32;
    int j = (int)(idx % 32);
    dtr_bf[idx] = (j < DTRANK) ? f2bf(xdbl[t * 64 + j]) : (unsigned short)0;
}

// ---------------------------------------------------------------------------
// In-place softplus (bias already added by GEMM)
// ---------------------------------------------------------------------------
__global__ void softplus_kernel(float* __restrict__ dt) {
    size_t idx = (size_t)blockIdx.x * blockDim.x + threadIdx.x;
    if (idx >= (size_t)TOK * DI) return;
    float v = dt[idx];
    dt[idx] = (v > 20.f) ? v : log1pf(__expf(v));
}

// ---------------------------------------------------------------------------
// Selective scan: one thread per (batch, channel), sequential over SEQ,
// 16 states. dir=1 scans right-to-left (== flip(ssm(flip(x))) output order).
// x_dbl row: [0..23]=dtr, [24..39]=B, [40..55]=C (row stride 64).
// ---------------------------------------------------------------------------
__global__ void scan_kernel(const float* __restrict__ u,
                            const float* __restrict__ dt,
                            const float* __restrict__ xdbl,
                            const float* __restrict__ A_log,
                            const float* __restrict__ D_skip,
                            unsigned short* __restrict__ y_bf,
                            int dir) {
    int idx = blockIdx.x * blockDim.x + threadIdx.x;
    if (idx >= BATCH * DI) return;
    int b = idx / DI, d = idx % DI;
    float A[DSTATE];
    #pragma unroll
    for (int n = 0; n < DSTATE; ++n) A[n] = -__expf(A_log[d * DSTATE + n]);
    const float Dk = D_skip[d];
    float h[DSTATE];
    #pragma unroll
    for (int n = 0; n < DSTATE; ++n) h[n] = 0.f;

    for (int s = 0; s < SEQ; ++s) {
        int t = dir ? (SEQ - 1 - s) : s;
        size_t tok = (size_t)b * SEQ + t;
        float dtv = dt[tok * DI + d];
        float uv  = u[tok * DI + d];
        const float* xd = xdbl + tok * 64;
        float y = 0.f;
        #pragma unroll
        for (int n = 0; n < DSTATE; ++n) {
            float dA = __expf(dtv * A[n]);
            h[n] = dA * h[n] + dtv * xd[24 + n] * uv;
            y += h[n] * xd[40 + n];
        }
        y += uv * Dk;
        y_bf[tok * DI + d] = f2bf(y);
    }
}

// ---------------------------------------------------------------------------
// Gate: ys = (h_fwd + h_bwd) * silu(zb)   -> bf16 for the final GEMM
// ---------------------------------------------------------------------------
__global__ void gate_kernel(const float* __restrict__ h0,
                            const float* __restrict__ h1,
                            const float* __restrict__ xz,
                            unsigned short* __restrict__ ys_bf) {
    size_t idx = (size_t)blockIdx.x * blockDim.x + threadIdx.x;
    if (idx >= (size_t)TOK * DM) return;
    size_t t = idx / DM;
    int c = (int)(idx % DM);
    float z = xz[t * (2 * DM) + DM + c];
    ys_bf[idx] = f2bf((h0[idx] + h1[idx]) * silu_f(z));
}

// ---------------------------------------------------------------------------
// Host-side launch
// ---------------------------------------------------------------------------
static inline dim3 g1(size_t n, int blk) { return dim3((unsigned)((n + blk - 1) / blk)); }

extern "C" void kernel_launch(void* const* d_in, const int* in_sizes, int n_in,
                              void* d_out, int out_size, void* d_ws, size_t ws_size,
                              hipStream_t stream) {
    const float* x      = (const float*)d_in[0];
    const float* ln_g   = (const float*)d_in[1];
    const float* ln_b   = (const float*)d_in[2];
    const float* W_in   = (const float*)d_in[3];
    const float* b_in   = (const float*)d_in[4];
    const float* W_si   = (const float*)d_in[5];
    const float* b_si   = (const float*)d_in[6];
    const float* conv_w = (const float*)d_in[7];
    const float* conv_b = (const float*)d_in[8];
    const float* W_x    = (const float*)d_in[9];
    const float* W_dt   = (const float*)d_in[10];
    const float* b_dt   = (const float*)d_in[11];
    const float* A_log  = (const float*)d_in[12];
    const float* D_skip = (const float*)d_in[13];
    const float* W_so   = (const float*)d_in[14];
    const float* b_so   = (const float*)d_in[15];
    const float* W_out  = (const float*)d_in[16];
    const float* b_out  = (const float*)d_in[17];
    float* out = (float*)d_out;

    // -------- workspace carve-out (256B aligned) --------
    char* wp = (char*)d_ws;
    auto carve = [&](size_t bytes) -> void* {
        void* p = (void*)wp;
        wp += (bytes + 255) & ~(size_t)255;
        return p;
    };
    unsigned short* wt_in  = (unsigned short*)carve((size_t)768 * 384 * 2);
    unsigned short* wt_si  = (unsigned short*)carve((size_t)768 * 384 * 2);
    unsigned short* wt_x   = (unsigned short*)carve((size_t)64  * 768 * 2);
    unsigned short* wt_dt  = (unsigned short*)carve((size_t)768 * 32  * 2);
    unsigned short* wt_so  = (unsigned short*)carve((size_t)384 * 768 * 2);
    unsigned short* wt_out = (unsigned short*)carve((size_t)384 * 384 * 2);
    unsigned short* xn_bf  = (unsigned short*)carve((size_t)TOK * DM * 2);
    float*          xz     = (float*)carve((size_t)TOK * 2 * DM * 4);
    unsigned short* xb_bf  = (unsigned short*)carve((size_t)TOK * DM * 2);
    float*          upre   = (float*)carve((size_t)TOK * DI * 4);
    float*          u_f    = (float*)carve((size_t)TOK * DI * 4);
    unsigned short* u_bf   = (unsigned short*)carve((size_t)TOK * DI * 2);
    float*          xdbl   = (float*)carve((size_t)TOK * 64 * 4);
    unsigned short* dtr_bf = (unsigned short*)carve((size_t)TOK * 32 * 2);
    float*          dt     = (float*)carve((size_t)TOK * DI * 4);
    unsigned short* y_bf   = (unsigned short*)carve((size_t)TOK * DI * 2);
    float*          h_dir0 = (float*)carve((size_t)TOK * DM * 4);
    float*          h_dir1 = (float*)carve((size_t)TOK * DM * 4);
    unsigned short* ys_bf  = (unsigned short*)carve((size_t)TOK * DM * 2);
    (void)ws_size; (void)n_in; (void)in_sizes; (void)out_size;

    // -------- weight transposes (f32 -> bf16, [N][K] with padding) --------
    convtrans_kernel<<<g1((size_t)768 * 384, 256), 256, 0, stream>>>(W_in,  wt_in,  384, 768, 384, 768);
    convtrans_kernel<<<g1((size_t)768 * 384, 256), 256, 0, stream>>>(W_si,  wt_si,  384, 768, 384, 768);
    convtrans_kernel<<<g1((size_t)64  * 768, 256), 256, 0, stream>>>(W_x,   wt_x,   768, 56,  768, 64);
    convtrans_kernel<<<g1((size_t)768 * 32,  256), 256, 0, stream>>>(W_dt,  wt_dt,  24,  768, 32,  768);
    convtrans_kernel<<<g1((size_t)384 * 768, 256), 256, 0, stream>>>(W_so,  wt_so,  768, 384, 768, 384);
    convtrans_kernel<<<g1((size_t)384 * 384, 256), 256, 0, stream>>>(W_out, wt_out, 384, 384, 384, 384);

    // -------- LayerNorm --------
    ln_kernel<<<dim3(TOK), 128, 0, stream>>>(x, ln_g, ln_b, xn_bf);

    // -------- xz = xn @ W_in + b_in   (4096x768, K=384) --------
    gemm_wmma_bf16<<<dim3(2 * DM / 64, TOK / 64), 128, 0, stream>>>(
        xn_bf, wt_in, b_in, nullptr, xz, TOK, 2 * DM, DM);

    // -------- split xb (bf16) --------
    split_kernel<<<g1((size_t)TOK * DM, 256), 256, 0, stream>>>(xz, xb_bf);

    // -------- u_pre = xb @ W_si + b_si   (4096x768, K=384) --------
    gemm_wmma_bf16<<<dim3(DI / 64, TOK / 64), 128, 0, stream>>>(
        xb_bf, wt_si, b_si, nullptr, upre, TOK, DI, DM);

    // -------- per-direction SSM --------
    for (int dir = 0; dir < 2; ++dir) {
        float* h_dir = dir ? h_dir1 : h_dir0;

        conv_silu_kernel<<<g1((size_t)TOK * DI, 256), 256, 0, stream>>>(
            upre, conv_w, conv_b, u_f, u_bf, dir);

        // x_dbl = u @ W_x   (4096x64, K=768; cols 56..63 are zero-padded)
        gemm_wmma_bf16<<<dim3(64 / 64, TOK / 64), 128, 0, stream>>>(
            u_bf, wt_x, nullptr, nullptr, xdbl, TOK, 64, DI);

        dtr_kernel<<<g1((size_t)TOK * 32, 256), 256, 0, stream>>>(xdbl, dtr_bf);

        // dt = dtr @ W_dt + b_dt   (4096x768, K=32 padded)
        gemm_wmma_bf16<<<dim3(DI / 64, TOK / 64), 128, 0, stream>>>(
            dtr_bf, wt_dt, b_dt, nullptr, dt, TOK, DI, 32);

        softplus_kernel<<<g1((size_t)TOK * DI, 256), 256, 0, stream>>>(dt);

        scan_kernel<<<g1((size_t)BATCH * DI, 256), 256, 0, stream>>>(
            u_f, dt, xdbl, A_log, D_skip, y_bf, dir);

        // h_dir = y @ W_so + b_so   (4096x384, K=768)
        gemm_wmma_bf16<<<dim3(DM / 64, TOK / 64), 128, 0, stream>>>(
            y_bf, wt_so, b_so, nullptr, h_dir, TOK, DM, DI);
    }

    // -------- gate + final projection with residual --------
    gate_kernel<<<g1((size_t)TOK * DM, 256), 256, 0, stream>>>(h_dir0, h_dir1, xz, ys_bf);

    gemm_wmma_bf16<<<dim3(DM / 64, TOK / 64), 128, 0, stream>>>(
        ys_bf, wt_out, b_out, x, out, TOK, DM, DM);
}
